// ExtraAttention_70136815944194
// MI455X (gfx1250) — compile-verified, matrix operand order checked by reading, not run on hardware
//
#include <hip/hip_runtime.h>
#include <hip/hip_bf16.h>
#include <math.h>

// ---- problem constants (from reference) ----
#define BATCH 4
#define SEQ   1024
#define PMEM  128
#define TTOT  (PMEM + SEQ)   // 1152
#define NXD   1024
#define HEADS 16
#define DH    64             // NXD / HEADS

typedef __bf16 bf16_t;
typedef bf16_t v16bf __attribute__((ext_vector_type(16)));
typedef float  v8f   __attribute__((ext_vector_type(8)));
typedef int    v4i   __attribute__((ext_vector_type(4)));

// ---------------- CDNA5 async global->LDS path ----------------
#if defined(__AMDGCN__) && __has_builtin(__builtin_amdgcn_global_load_async_to_lds_b128)
#define HAVE_ASYNC_LDS 1
#else
#define HAVE_ASYNC_LDS 0
#endif

// copy 16B global -> LDS; async (ASYNCcnt) when available, sync fallback else.
// builtin signature (from clang diagnostic): (v4i AS(1)*, v4i AS(3)*, Ii, Ii)
__device__ __forceinline__ void async_ld16(const bf16_t* g, bf16_t* l) {
#if HAVE_ASYNC_LDS
  __builtin_amdgcn_global_load_async_to_lds_b128(
      (__attribute__((address_space(1))) v4i*)g,
      (__attribute__((address_space(3))) v4i*)l, 0, 0);
#else
  *(uint4*)l = *(const uint4*)g;
#endif
}
__device__ __forceinline__ void wait_async0() {
#if HAVE_ASYNC_LDS
#if __has_builtin(__builtin_amdgcn_s_wait_asynccnt)
  __builtin_amdgcn_s_wait_asynccnt(0);
#else
  asm volatile("s_wait_asynccnt 0" ::: "memory");
#endif
#endif
}

// ---------------- WMMA helpers ----------------
__device__ __forceinline__ v8f wmma_bf16(v16bf a, v16bf b, v8f c) {
  // (neg_a, A, neg_b, B, c_mod, C, reuse_a, reuse_b)
  return __builtin_amdgcn_wmma_f32_16x16x32_bf16(false, a, false, b, (short)0, c,
                                                 false, false);
}

// 16x32 bf16 fragment, contiguous per-lane rows: lane holds row (lane&15);
// element e maps to K = (e/8)*16 + (lane/16)*8 + (e%8) + k0.
// Used for A (rows of A) and for B when LDS holds rows of B^T.
// Lowers to 2x ds_load_b128 per fragment.
__device__ __forceinline__ v16bf load_frag_rows(const bf16_t* p, int ld, int k0) {
  const int lane = threadIdx.x & 31;
  const bf16_t* r = p + (size_t)(lane & 15) * ld;
  const int kb = ((lane >> 4) << 3) + k0;
  v16bf f;
#pragma unroll
  for (int e = 0; e < 8; ++e) f[e] = r[kb + e];
#pragma unroll
  for (int e = 0; e < 8; ++e) f[e + 8] = r[kb + 16 + e];
  return f;
}

// Cross-lane reductions within each 16-lane half of a wave32 (C-fragment rows
// live across lanes 0..15 / 16..31 of one half).
__device__ __forceinline__ float lane16_max(float v) {
  const int lane = threadIdx.x & 31;
#pragma unroll
  for (int m = 1; m < 16; m <<= 1) {
    int o = __builtin_amdgcn_ds_bpermute(((lane ^ m) << 2), __float_as_int(v));
    v = fmaxf(v, __int_as_float(o));
  }
  return v;
}
__device__ __forceinline__ float lane16_sum(float v) {
  const int lane = threadIdx.x & 31;
#pragma unroll
  for (int m = 1; m < 16; m <<= 1) {
    int o = __builtin_amdgcn_ds_bpermute(((lane ^ m) << 2), __float_as_int(v));
    v += __int_as_float(o);
  }
  return v;
}

// ---------------- elementwise kernels ----------------
__global__ void cvt_f32_bf16(const float* __restrict__ in, bf16_t* __restrict__ out,
                             int n) {
  int i = blockIdx.x * blockDim.x + threadIdx.x;
  if (i < n) out[i] = (bf16_t)in[i];
}

// QKV [4096,3072] -> Q [B,S,NX], K/V self parts (rows P..P+S-1 of [B,T,NX])
__global__ void scatter_qkv(const bf16_t* __restrict__ QKV, bf16_t* __restrict__ Q,
                            bf16_t* __restrict__ Kc, bf16_t* __restrict__ Vc) {
  int i = blockIdx.x * blockDim.x + threadIdx.x;   // < 4096*3072
  int row = i / (3 * NXD);
  int col = i - row * (3 * NXD);
  int part = col >> 10, c = col & (NXD - 1);
  int b = row >> 10, s = row & (SEQ - 1);
  bf16_t v = QKV[i];
  if (part == 0)      Q[(size_t)row * NXD + c] = v;
  else if (part == 1) Kc[((size_t)(b * TTOT + PMEM + s)) * NXD + c] = v;
  else                Vc[((size_t)(b * TTOT + PMEM + s)) * NXD + c] = v;
}

// EKV [512,2048] -> K/V memory parts (rows 0..P-1 of [B,T,NX])
__global__ void scatter_ekv(const bf16_t* __restrict__ EKV, bf16_t* __restrict__ Kc,
                            bf16_t* __restrict__ Vc) {
  int i = blockIdx.x * blockDim.x + threadIdx.x;   // < 512*2048
  int row = i >> 11;
  int col = i & 2047;
  int part = col >> 10, c = col & (NXD - 1);
  int b = row >> 7, p = row & (PMEM - 1);
  bf16_t v = EKV[i];
  if (part == 0) Kc[((size_t)(b * TTOT + p)) * NXD + c] = v;
  else           Vc[((size_t)(b * TTOT + p)) * NXD + c] = v;
}

// ---------------- generic bf16 WMMA GEMM: C = A[MxK] * B[KxN] + bias ----------------
// 128 threads = 4 waves in 2x2 -> 64x64 C tile; each wave owns a 32x32 quadrant
// (2 A-frags x 2 B-frags -> 4 wmma per 32-wide K chunk).  Double-buffered LDS:
// A staged with async global->LDS (ASYNCcnt), B^T staged via registers
// (load-early / transpose-store-late), one barrier per K chunk.
#define GLD 40   // padded LDS row stride (bf16 elems); 80B = 5x16B, keeps 16B align
__global__ void gemm_bf16(const bf16_t* __restrict__ A, const bf16_t* __restrict__ B,
                          const float* __restrict__ bias, bf16_t* __restrict__ Cb,
                          float* __restrict__ Cf, int M, int N, int K) {
  __shared__ bf16_t As[2][64 * GLD];   // [m 0..63][k 0..31]
  __shared__ bf16_t BsT[2][64 * GLD];  // [n 0..63][k 0..31]
  const int tid  = threadIdx.x;
  const int lane = tid & 31;
  const int wv   = tid >> 5;
  const int m0 = blockIdx.y * 64;
  const int n0 = blockIdx.x * 64;
  const int mw = (wv >> 1) << 5;    // 0 or 32
  const int nw = (wv & 1) << 5;     // 0 or 32

  // staging coordinates
  const int arow = tid >> 1;          // 0..63
  const int acol = (tid & 1) << 4;    // 0 or 16
  const int bkr  = (tid * 16) >> 6;   // 0..31 (k row)
  const int bnc  = (tid * 16) & 63;   // 0,16,32,48 (n col)

  bf16_t btmp[16];
  auto stage_A = [&](int kk, int buf) {
    const bf16_t* src = &A[(size_t)(m0 + arow) * K + kk + acol];
    bf16_t* dst = &As[buf][arow * GLD + acol];
    async_ld16(src, dst);
    async_ld16(src + 8, dst + 8);
  };
  auto load_B = [&](int kk) {
    const bf16_t* src = &B[(size_t)(kk + bkr) * N + n0 + bnc];
#pragma unroll
    for (int e = 0; e < 16; ++e) btmp[e] = src[e];
  };
  auto store_B = [&](int buf) {
#pragma unroll
    for (int e = 0; e < 16; ++e) BsT[buf][(bnc + e) * GLD + bkr] = btmp[e];
  };

  v8f acc00 = {}, acc01 = {}, acc10 = {}, acc11 = {};
  stage_A(0, 0);
  load_B(0);
  store_B(0);
  wait_async0();
  __syncthreads();

  const int niter = K >> 5;
  for (int it = 0; it < niter; ++it) {
    const int cur = it & 1, nxt = cur ^ 1;
    const int kk = it << 5;
    const bool has_next = (it + 1 < niter);
    if (has_next) {             // issue next-chunk copies before compute
      stage_A(kk + 32, nxt);
      load_B(kk + 32);
      if (it + 2 < niter)       // global_prefetch_b8 for chunk after next
        __builtin_prefetch(&A[(size_t)(m0 + arow) * K + kk + 64], 0, 3);
    }
    v16bf a0 = load_frag_rows(&As[cur][(mw + 0) * GLD], GLD, 0);
    v16bf a1 = load_frag_rows(&As[cur][(mw + 16) * GLD], GLD, 0);
    v16bf b0 = load_frag_rows(&BsT[cur][(nw + 0) * GLD], GLD, 0);
    v16bf b1 = load_frag_rows(&BsT[cur][(nw + 16) * GLD], GLD, 0);
    acc00 = wmma_bf16(a0, b0, acc00);
    acc01 = wmma_bf16(a0, b1, acc01);
    acc10 = wmma_bf16(a1, b0, acc10);
    acc11 = wmma_bf16(a1, b1, acc11);
    if (has_next) store_B(nxt); // transpose-store after compute (hides latency)
    wait_async0();
    __syncthreads();
  }

  const int hi = lane >> 4;
  v8f* accs[2][2] = {{&acc00, &acc01}, {&acc10, &acc11}};
#pragma unroll
  for (int i = 0; i < 2; ++i) {
#pragma unroll
    for (int j = 0; j < 2; ++j) {
      int n = n0 + nw + j * 16 + (lane & 15);
      float bn = bias ? bias[n] : 0.f;
#pragma unroll
      for (int r = 0; r < 8; ++r) {
        int m = m0 + mw + i * 16 + r + (hi << 3);
        float v = (*accs[i][j])[r] + bn;
        if (Cf) Cf[(size_t)m * N + n] = v;
        else    Cb[(size_t)m * N + n] = (bf16_t)v;
      }
    }
  }
}

__device__ __forceinline__ float apply_mask(float v, int t, int qrow, int b,
                                            const float* __restrict__ Mmask,
                                            const float* __restrict__ amask) {
  if (t < PMEM) return v + (1.f - Mmask[b * PMEM + t]) * -10000.f;
  int ts = t - PMEM;
  float am = amask[b * SEQ + ts];
  return (ts <= qrow ? v : -10000.f) + am;
}

// ---------------- flash attention over concatenated [mem | self] keys ----------------
// grid: B*H*(S/64) blocks, 128 threads (4 waves, 16 query rows each).
// Double-buffered K/V tiles; K async global->LDS, V transposed via registers.
// P tile is per-wave (LDS ops within a wave are in-order) -> no mid-loop barrier.
#define LDK 72   // K tile row stride: 32 keys x 64 dh (padded; 144B = 9x16B)
#define LDV 40   // V^T tile row stride: 64 dh rows x 32 keys (padded)
#define LDP 40   // P tile row stride: 16 rows x 32 keys (padded)
__global__ void attn_kernel(const bf16_t* __restrict__ Q, const bf16_t* __restrict__ Kc,
                            const bf16_t* __restrict__ Vc,
                            const float* __restrict__ Mmask,
                            const float* __restrict__ amask,
                            bf16_t* __restrict__ Ah) {
  __shared__ bf16_t Kt[2][32 * LDK];    // [key 0..31][dh 0..63]
  __shared__ bf16_t VtT[2][64 * LDV];   // [dh 0..63][key 0..31]
  __shared__ bf16_t Pt[4][16 * LDP];    // per-wave exp'd scores [qrow][key]
  const int tid  = threadIdx.x;
  const int lane = tid & 31;
  const int wv   = tid >> 5;
  const int bid  = blockIdx.x;
  const int b  = bid >> 8;
  const int h  = (bid >> 4) & 15;
  const int qt = bid & 15;
  const int qbase = qt * 64 + wv * 16;

  // staging coordinates: each thread moves 16 contiguous dh elems of one key row
  const int rr = tid >> 2;            // key row 0..31
  const int cc = (tid & 3) << 4;      // dh col 0,16,32,48

  bf16_t vtmp[16];
  auto stage_K = [&](int j, int buf) {
    const bf16_t* src = &Kc[((size_t)(b * TTOT + j + rr) * NXD) + h * DH + cc];
    bf16_t* dst = &Kt[buf][rr * LDK + cc];
    async_ld16(src, dst);
    async_ld16(src + 8, dst + 8);
  };
  auto load_V = [&](int j) {
    const bf16_t* src = &Vc[((size_t)(b * TTOT + j + rr) * NXD) + h * DH + cc];
#pragma unroll
    for (int e = 0; e < 16; ++e) vtmp[e] = src[e];
  };
  auto store_V = [&](int buf) {
#pragma unroll
    for (int e = 0; e < 16; ++e) VtT[buf][(cc + e) * LDV + rr] = vtmp[e];
  };

  // Q fragments for this wave's 16 rows (dh = 64 -> two 16x32 A-fragments)
  const bf16_t* qp = Q + ((size_t)(b * SEQ + qbase) * NXD) + h * DH;
  const v16bf qa0 = load_frag_rows(qp, NXD, 0);
  const v16bf qa1 = load_frag_rows(qp, NXD, 32);

  const int hi = lane >> 4;
  float mrow[8], lrow[8];
  v8f o0 = {}, o1 = {}, o2 = {}, o3 = {};
#pragma unroll
  for (int r = 0; r < 8; ++r) { mrow[r] = -3.0e38f; lrow[r] = 0.f; }
  const float rscale = 0.125f;  // 1/sqrt(dh)

  stage_K(0, 0);
  load_V(0);
  store_V(0);
  wait_async0();
  __syncthreads();

  const int niter = TTOT / 32;   // 36
  for (int it = 0; it < niter; ++it) {
    const int j = it << 5;
    const int cur = it & 1, nxt = cur ^ 1;
    const bool has_next = (it + 1 < niter);
    if (has_next) {
      stage_K(j + 32, nxt);
      load_V(j + 32);
      if (it + 2 < niter)
        __builtin_prefetch(&Kc[((size_t)(b * TTOT + j + 64 + rr) * NXD) + h * DH], 0, 3);
    }

    // scores: 16 q-rows x 32 keys
    v8f s0 = {}, s1 = {};
    s0 = wmma_bf16(qa0, load_frag_rows(&Kt[cur][0], LDK, 0), s0);
    s0 = wmma_bf16(qa1, load_frag_rows(&Kt[cur][0], LDK, 32), s0);
    s1 = wmma_bf16(qa0, load_frag_rows(&Kt[cur][16 * LDK], LDK, 0), s1);
    s1 = wmma_bf16(qa1, load_frag_rows(&Kt[cur][16 * LDK], LDK, 32), s1);

    // masks + online softmax (rows distributed across 16-lane halves)
#pragma unroll
    for (int r = 0; r < 8; ++r) {
      int qm = r + (hi << 3);
      int qrow = qbase + qm;
      int t0 = j + (lane & 15);
      float v0 = apply_mask(s0[r] * rscale, t0, qrow, b, Mmask, amask);
      float v1 = apply_mask(s1[r] * rscale, t0 + 16, qrow, b, Mmask, amask);
      float mx = lane16_max(fmaxf(v0, v1));
      float mn = fmaxf(mrow[r], mx);
      float alpha = expf(mrow[r] - mn);
      float p0 = expf(v0 - mn);
      float p1 = expf(v1 - mn);
      lrow[r] = lrow[r] * alpha + lane16_sum(p0 + p1);
      mrow[r] = mn;
      o0[r] *= alpha; o1[r] *= alpha; o2[r] *= alpha; o3[r] *= alpha;
      Pt[wv][qm * LDP + (lane & 15)]      = (bf16_t)p0;
      Pt[wv][qm * LDP + 16 + (lane & 15)] = (bf16_t)p1;
    }

    // O += P(16x32) * V(32x64); Pt is per-wave, LDS ops in-order within a wave
    v16bf pa = load_frag_rows(Pt[wv], LDP, 0);
    o0 = wmma_bf16(pa, load_frag_rows(&VtT[cur][0 * LDV],  LDV, 0), o0);
    o1 = wmma_bf16(pa, load_frag_rows(&VtT[cur][16 * LDV], LDV, 0), o1);
    o2 = wmma_bf16(pa, load_frag_rows(&VtT[cur][32 * LDV], LDV, 0), o2);
    o3 = wmma_bf16(pa, load_frag_rows(&VtT[cur][48 * LDV], LDV, 0), o3);

    if (has_next) store_V(nxt);
    wait_async0();
    __syncthreads();
  }

  // normalize + merge heads into Ah [B,S,NX]
#pragma unroll
  for (int r = 0; r < 8; ++r) {
    int qm = r + (hi << 3);
    float inv = 1.f / lrow[r];
    bf16_t* op = Ah + (size_t)(b * SEQ + qbase + qm) * NXD + h * DH + (lane & 15);
    op[0]  = (bf16_t)(o0[r] * inv);
    op[16] = (bf16_t)(o1[r] * inv);
    op[32] = (bf16_t)(o2[r] * inv);
    op[48] = (bf16_t)(o3[r] * inv);
  }
}

// ---------------- host launcher ----------------
extern "C" void kernel_launch(void* const* d_in, const int* in_sizes, int n_in,
                              void* d_out, int out_size, void* d_ws, size_t ws_size,
                              hipStream_t stream) {
  const float* x     = (const float*)d_in[0];
  const float* M     = (const float*)d_in[1];
  const float* Mmask = (const float*)d_in[2];
  const float* amask = (const float*)d_in[3];
  const float* W_attn = (const float*)d_in[4];
  const float* b_attn = (const float*)d_in[5];
  const float* W_mem  = (const float*)d_in[6];
  const float* b_mem  = (const float*)d_in[7];
  const float* W_proj = (const float*)d_in[8];
  const float* b_proj = (const float*)d_in[9];
  float* out = (float*)d_out;

  // workspace carve-up (~85 MB, bf16 staging buffers)
  char* ws = (char*)d_ws;
  size_t off = 0;
  auto carve = [&](size_t elems) {
    bf16_t* p = (bf16_t*)(ws + off);
    off += (elems * sizeof(bf16_t) + 255) & ~(size_t)255;
    return p;
  };
  bf16_t* Xh  = carve((size_t)BATCH * SEQ * NXD);
  bf16_t* Mh  = carve((size_t)BATCH * PMEM * NXD);
  bf16_t* Wah = carve((size_t)NXD * 3 * NXD);
  bf16_t* Wmh = carve((size_t)NXD * 2 * NXD);
  bf16_t* Wph = carve((size_t)NXD * NXD);
  bf16_t* QKV = carve((size_t)BATCH * SEQ * 3 * NXD);
  bf16_t* EKV = carve((size_t)BATCH * PMEM * 2 * NXD);
  bf16_t* Qh  = carve((size_t)BATCH * SEQ * NXD);
  bf16_t* Kc  = carve((size_t)BATCH * TTOT * NXD);
  bf16_t* Vc  = carve((size_t)BATCH * TTOT * NXD);
  bf16_t* Ah  = carve((size_t)BATCH * SEQ * NXD);
  (void)ws_size; (void)n_in; (void)in_sizes; (void)out_size;

  auto cvt = [&](const float* src, bf16_t* dst, int n) {
    cvt_f32_bf16<<<(n + 255) / 256, 256, 0, stream>>>(src, dst, n);
  };
  cvt(x, Xh, BATCH * SEQ * NXD);
  cvt(M, Mh, BATCH * PMEM * NXD);
  cvt(W_attn, Wah, NXD * 3 * NXD);
  cvt(W_mem, Wmh, NXD * 2 * NXD);
  cvt(W_proj, Wph, NXD * NXD);

  // QKV = x @ W_attn + b_attn   (M=4096, N=3072, K=1024)
  gemm_bf16<<<dim3(3 * NXD / 64, BATCH * SEQ / 64), 128, 0, stream>>>(
      Xh, Wah, b_attn, QKV, nullptr, BATCH * SEQ, 3 * NXD, NXD);
  // EKV = M @ W_mem + b_mem     (M=512, N=2048, K=1024)
  gemm_bf16<<<dim3(2 * NXD / 64, BATCH * PMEM / 64), 128, 0, stream>>>(
      Mh, Wmh, b_mem, EKV, nullptr, BATCH * PMEM, 2 * NXD, NXD);

  scatter_qkv<<<(BATCH * SEQ * 3 * NXD) / 256, 256, 0, stream>>>(QKV, Qh, Kc, Vc);
  scatter_ekv<<<(BATCH * PMEM * 2 * NXD) / 256, 256, 0, stream>>>(EKV, Kc, Vc);

  // flash attention -> Ah [B,S,NX]
  attn_kernel<<<BATCH * HEADS * (SEQ / 64), 128, 0, stream>>>(Qh, Kc, Vc, Mmask,
                                                              amask, Ah);

  // out = Ah @ W_proj + b_proj  (f32 epilogue straight to d_out)
  gemm_bf16<<<dim3(NXD / 64, BATCH * SEQ / 64), 128, 0, stream>>>(
      Ah, Wph, b_proj, nullptr, out, BATCH * SEQ, NXD, NXD);
}
